// VDModule_26809185861900
// MI455X (gfx1250) — matrix operand order repaired and checked by reading, not compiled, not run on hardware
//
#include <hip/hip_runtime.h>
#include <stdint.h>

// ---------------------------------------------------------------------------
// Problem constants (from reference): B=16, N=8192, CF=256, COUT=64
// feat = W(64x259) @ concat(xyz^T, features)(259x8192)  per batch  (bf16 WMMA)
// then per-batch argsort of dist-to-last-point, gather columns, maxpool by 4.
// Memory-bound: ~210 MB total traffic -> ~9us at 23.3 TB/s.
//
// Virtual K layout (branchless): kk 0..255  = features rows 0..255
//                                kk 256..258 = xyz components 0..2
//                                kk 259..287 = zero pad
// LDS W is staged with the same permutation, so the product is unchanged.
// ---------------------------------------------------------------------------
#define B_    16
#define N_    8192
#define CF_   256
#define KTOT  259        // 3 (xyz) + 256 (features)
#define KPAD  288        // 9 k-tiles of 32
#define COUT_ 64
#define NT    128        // columns per block (8 waves x 16)
#define NOUT  (N_ / 4)   // 2048

typedef __attribute__((ext_vector_type(16))) __bf16 v16bf;
typedef __attribute__((ext_vector_type(8)))  float  v8f;

union FragBF { v16bf h; unsigned u[8]; };

// float -> bf16 bits, round-to-nearest-even (used only for one-time W staging)
__device__ __forceinline__ unsigned short f2bf(float f) {
    union { float f; unsigned u; } in; in.f = f;
    unsigned u = in.u;
    return (unsigned short)((u + 0x7FFFu + ((u >> 16) & 1u)) >> 16);
}

// Pack two floats into two bf16 (round-half-up) with a single v_perm_b32.
// sel 0x07060302: dst bytes {a.b2, a.b3, b.b2, b.b3}  (a = low half / K, b = K+1)
__device__ __forceinline__ unsigned pack_bf16(float a, float b) {
    union { float f; unsigned u; } ua, ub; ua.f = a; ub.f = b;
    return __builtin_amdgcn_perm(ub.u + 0x8000u, ua.u + 0x8000u, 0x07060302u);
}

// W element under the permuted-K layout
__device__ __forceinline__ float w_perm(const float* __restrict__ W, int row, int kk) {
    if (kk < CF_)       return W[row * KTOT + kk + 3];   // features part
    else if (kk < KTOT) return W[row * KTOT + kk - CF_]; // xyz part
    else                return 0.0f;                     // pad
}

// ---------------------------------------------------------------------------
// Kernel 1: feat[b, 0:64, n] = W @ concat(xyz^T, features)   via WMMA bf16
// grid (N/128, B), block 256 (8 waves). Wave w: columns [w*16, w*16+16),
// all 4 M-tiles (M=64). K loop: 8 regular feature tiles + 1 xyz tile.
// ---------------------------------------------------------------------------
__global__ __launch_bounds__(256)
void gemm_feat_kernel(const float* __restrict__ xyz,
                      const float* __restrict__ features,
                      const float* __restrict__ W,
                      float* __restrict__ feat) {
    // W staged as permuted, zero-padded bf16: row-major 64 x 288, packed pairs.
    __shared__ unsigned Wsh[COUT_ * KPAD / 2];   // 36,864 B LDS

    const int tid = threadIdx.x;
    const int b   = blockIdx.y;

    for (int i = tid; i < COUT_ * KPAD / 2; i += 256) {
        int row = i / (KPAD / 2);
        int k0  = (i % (KPAD / 2)) * 2;
        Wsh[i] = (unsigned)f2bf(w_perm(W, row, k0)) |
                 ((unsigned)f2bf(w_perm(W, row, k0 + 1)) << 16);
    }
    __syncthreads();

    const int wave = tid >> 5;
    const int lane = tid & 31;
    const int hlf  = lane >> 4;                          // half-wave select
    const int l16  = lane & 15;
    const int ncol = blockIdx.x * NT + wave * 16 + l16;  // B/C column (N index)

    // Per-column feature pointer: row r -> fcol[r * N_]
    const float* __restrict__ fcol = features + (size_t)b * CF_ * N_ + ncol;

    v8f acc[4];
    #pragma unroll
    for (int m = 0; m < 4; ++m)
        acc[m] = (v8f){0.f, 0.f, 0.f, 0.f, 0.f, 0.f, 0.f, 0.f};

    // A-fragment LDS addressing (16x32 16-bit layout):
    // lanes 0-15 K base 0, lanes 16-31 K base 8; VGPR j<4 -> K=base+2j,
    // j>=4 -> K=16+base+2(j-4).  kk is always even -> packed-pair index.
    #define A_FRAG(af, m, kt)                                                  \
        {                                                                      \
            const int row_ = (m) * 16 + l16;                                   \
            _Pragma("unroll")                                                  \
            for (int j = 0; j < 8; ++j) {                                      \
                int kk = (kt) * 32 + hlf * 8 +                                 \
                         ((j < 4) ? 2 * j : 16 + 2 * (j - 4));                 \
                (af).u[j] = Wsh[(row_ * KPAD + kk) >> 1];                      \
            }                                                                  \
        }

    // ---- 8 fully regular feature k-tiles (branchless loads) ----
    #pragma unroll
    for (int kt = 0; kt < 8; ++kt) {
        if (kt + 1 < 8)
            __builtin_prefetch(&fcol[(size_t)((kt + 1) * 32) * N_], 0, 1);

        // B fragment: 32x16 16-bit, same K order as A:
        // K base 8*hlf; VGPR j<4 -> K=base+2j, j>=4 -> K=16+base+2(j-4).
        FragBF bf;
        #pragma unroll
        for (int j = 0; j < 8; ++j) {
            int r = kt * 32 + hlf * 8 + ((j < 4) ? 2 * j : 16 + 2 * (j - 4));          // feature row
            float f0 = fcol[(size_t)r * N_];
            float f1 = fcol[(size_t)(r + 1) * N_];
            bf.u[j] = pack_bf16(f0, f1);
        }

        #pragma unroll
        for (int m = 0; m < 4; ++m) {
            FragBF af;
            A_FRAG(af, m, kt)
            acc[m] = __builtin_amdgcn_wmma_f32_16x16x32_bf16(
                false, af.h, false, bf.h, (short)0, acc[m], false, false);
        }
    }

    // ---- k-tile 8: xyz contribution (kk 256..258), compile-time sparse B ----
    {
        const float x0 = xyz[((size_t)b * N_ + ncol) * 3 + 0];
        const float x1 = xyz[((size_t)b * N_ + ncol) * 3 + 1];
        const float x2 = xyz[((size_t)b * N_ + ncol) * 3 + 2];

        FragBF bx;
        // hlf==0 lanes carry K=256..258 in VGPRs 0-1; everything else is zero.
        bx.u[0] = hlf ? 0u : pack_bf16(x0, x1);
        bx.u[1] = hlf ? 0u : pack_bf16(x2, 0.0f);
        #pragma unroll
        for (int j = 2; j < 8; ++j) bx.u[j] = 0u;

        #pragma unroll
        for (int m = 0; m < 4; ++m) {
            FragBF af;
            A_FRAG(af, m, 8)
            acc[m] = __builtin_amdgcn_wmma_f32_16x16x32_bf16(
                false, af.h, false, bx.h, (short)0, acc[m], false, false);
        }
    }
    #undef A_FRAG

    // C/D layout: VGPR v, lanes 0-15: M=v, N=lane ; lanes 16-31: M=v+8.
    #pragma unroll
    for (int m = 0; m < 4; ++m) {
        #pragma unroll
        for (int v = 0; v < 8; ++v) {
            int row = m * 16 + v + hlf * 8;
            feat[((size_t)b * COUT_ + row) * N_ + ncol] = acc[m][v];
        }
    }
}

// ---------------------------------------------------------------------------
// Kernel 2: per-batch argsort of dist(new_xyz, new_xyz[N-1]) via bitonic sort
// in LDS. 1024 threads, 8 elems/thread; 48 KB LDS (f32 keys + u16 indices).
// ---------------------------------------------------------------------------
__global__ __launch_bounds__(1024)
void argsort_kernel(const float* __restrict__ new_xyz,
                    int* __restrict__ idx_out) {
    __shared__ float          skey[N_];   // 32 KB
    __shared__ unsigned short sidx[N_];   // 16 KB

    const int b   = blockIdx.x;
    const int tid = threadIdx.x;

    const float cx = new_xyz[((size_t)b * N_ + (N_ - 1)) * 3 + 0];
    const float cy = new_xyz[((size_t)b * N_ + (N_ - 1)) * 3 + 1];
    const float cz = new_xyz[((size_t)b * N_ + (N_ - 1)) * 3 + 2];

    for (int n = tid; n < N_; n += 1024) {
        float dx = new_xyz[((size_t)b * N_ + n) * 3 + 0] - cx;
        float dy = new_xyz[((size_t)b * N_ + n) * 3 + 1] - cy;
        float dz = new_xyz[((size_t)b * N_ + n) * 3 + 2] - cz;
        skey[n] = dx * dx + dy * dy + dz * dz;
        sidx[n] = (unsigned short)n;
    }
    __syncthreads();

    for (int k = 2; k <= N_; k <<= 1) {
        for (int j = k >> 1; j > 0; j >>= 1) {
            #pragma unroll
            for (int t = 0; t < N_ / 1024; ++t) {
                int i   = tid + t * 1024;
                int ixj = i ^ j;
                if (ixj > i) {                    // each pair handled once
                    bool up  = ((i & k) == 0);    // ascending region
                    float ki = skey[i], kj = skey[ixj];
                    if ((ki > kj) == up) {
                        skey[i] = kj; skey[ixj] = ki;
                        unsigned short ti = sidx[i];
                        sidx[i] = sidx[ixj]; sidx[ixj] = ti;
                    }
                }
            }
            __syncthreads();
        }
    }

    for (int n = tid; n < N_; n += 1024)
        idx_out[b * N_ + n] = (int)sidx[n];
}

// ---------------------------------------------------------------------------
// Kernel 3: gather sorted columns + maxpool groups of 4.
// One block per (o, b); feat row (32 KB) stays L0/L2 resident for the gather.
// ---------------------------------------------------------------------------
__global__ __launch_bounds__(256)
void pool_kernel(const float* __restrict__ feat,
                 const int* __restrict__ idx,
                 float* __restrict__ out) {
    const int o = blockIdx.x;
    const int b = blockIdx.y;
    const float* __restrict__ frow = feat + ((size_t)b * COUT_ + o) * N_;
    const int*   __restrict__ irow = idx + b * N_;

    for (int j = threadIdx.x; j < NOUT; j += 256) {
        float m0 = frow[irow[4 * j + 0]];
        float m1 = frow[irow[4 * j + 1]];
        float m2 = frow[irow[4 * j + 2]];
        float m3 = frow[irow[4 * j + 3]];
        out[((size_t)b * COUT_ + o) * NOUT + j] =
            fmaxf(fmaxf(m0, m1), fmaxf(m2, m3));
    }
}

// ---------------------------------------------------------------------------
extern "C" void kernel_launch(void* const* d_in, const int* in_sizes, int n_in,
                              void* d_out, int out_size, void* d_ws, size_t ws_size,
                              hipStream_t stream) {
    const float* xyz      = (const float*)d_in[0];   // (B, N, 3)
    const float* new_xyz  = (const float*)d_in[1];   // (B, N, 3)
    const float* features = (const float*)d_in[2];   // (B, CF, N)
    const float* W        = (const float*)d_in[3];   // (COUT, 3+CF)
    float* out = (float*)d_out;                      // (B, COUT, N/4)

    float* feat = (float*)d_ws;                                    // 32 MB
    int*   idxw = (int*)((char*)d_ws + (size_t)B_ * COUT_ * N_ * sizeof(float));

    gemm_feat_kernel<<<dim3(N_ / NT, B_), 256, 0, stream>>>(xyz, features, W, feat);
    argsort_kernel<<<dim3(B_), 1024, 0, stream>>>(new_xyz, idxw);
    pool_kernel<<<dim3(COUT_, B_), 256, 0, stream>>>(feat, idxw, out);
}
